// BertAttention_66116726555166
// MI455X (gfx1250) — matrix operand order repaired and checked by reading, not compiled, not run on hardware
//
#include <hip/hip_runtime.h>
#include <hip/hip_bf16.h>

typedef __attribute__((ext_vector_type(16))) __bf16 v16bf;
typedef __attribute__((ext_vector_type(8)))  float  v8f;
typedef __attribute__((ext_vector_type(4)))  int    v4i;

union ABfrag { v16bf v; uint4 q[2]; };

constexpr int Bsz = 4, Ssz = 2048, Hsz = 1024, NHD = 16, HDm = 64;
constexpr int Mrows = Bsz * Ssz; // 8192

static __device__ inline v8f vzero() {
  v8f z = {0.f,0.f,0.f,0.f,0.f,0.f,0.f,0.f};
  return z;
}

static __device__ inline v8f bwmma(v16bf a, v16bf b, v8f c) {
  // D = A(16x32 bf16) * B(32x16 bf16) + C(16x16 f32)
  return __builtin_amdgcn_wmma_f32_16x16x32_bf16(
      false, a, false, b, (short)0, c, false, false);
}

// --- async global->LDS copy (ASYNCcnt path) with sync fallback -------------
#if __has_builtin(__builtin_amdgcn_global_load_async_to_lds_b128)
#define ASYNC_LDS 1
typedef __attribute__((address_space(1))) v4i* gv4i_p;  // global int4*
typedef __attribute__((address_space(3))) v4i* lv4i_p;  // LDS int4*
#endif

static __device__ inline void cp16_lds(__bf16* dst, const __bf16* src) {
#ifdef ASYNC_LDS
  __builtin_amdgcn_global_load_async_to_lds_b128(
      (gv4i_p)(src), (lv4i_p)(dst), 0, 0);
#else
  *(uint4*)dst = *(const uint4*)src;
#endif
}

static __device__ inline void cp_wait() {
#ifdef ASYNC_LDS
#if __has_builtin(__builtin_amdgcn_s_wait_asynccnt)
  __builtin_amdgcn_s_wait_asynccnt(0);
#else
  asm volatile("s_wait_asynccnt 0x0" ::: "memory");
#endif
#endif
}

// ---------------------------------------------------------------------------
// Kernel 1: QKV projection.  grid = (M/128, N/128, 3), block = 256 (8 waves).
// C[m,n] = X[m,:] @ W[:,n] + bias[n], stored bf16 into [B,NH,S,HD].
// ---------------------------------------------------------------------------
__global__ __launch_bounds__(256) void qkv_gemm(
    const float* __restrict__ X,
    const float* __restrict__ Wq, const float* __restrict__ bq,
    const float* __restrict__ Wk, const float* __restrict__ bk,
    const float* __restrict__ Wv, const float* __restrict__ bv,
    __bf16* __restrict__ Qb, __bf16* __restrict__ Kb, __bf16* __restrict__ Vb)
{
  const int which = blockIdx.z;
  const float* W    = (which == 0) ? Wq : (which == 1) ? Wk : Wv;
  const float* bias = (which == 0) ? bq : (which == 1) ? bk : bv;
  __bf16* Out       = (which == 0) ? Qb : (which == 1) ? Kb : Vb;

  __shared__ __align__(16) __bf16 As[128 * 32];  // [m][k]
  __shared__ __align__(16) __bf16 Bs[128 * 32];  // [n][k] (transposed)

  const int tid  = threadIdx.x;
  const int wave = tid >> 5;
  const int lane = tid & 31;
  const int hl   = lane >> 4;   // lane half (0/1)
  const int l16  = lane & 15;
  const int m0   = blockIdx.x * 128;
  const int n0   = blockIdx.y * 128;

  v8f acc[8];
  #pragma unroll
  for (int t = 0; t < 8; ++t) acc[t] = vzero();

  for (int k0 = 0; k0 < Hsz; k0 += 32) {
    // --- stage A tile: 128x32 f32 -> bf16, row-major, coalesced reads
    {
      const int row = tid >> 1;
      const int col = (tid & 1) * 16;
      const float* src = X + (size_t)(m0 + row) * Hsz + k0 + col;
      __builtin_prefetch(src + 32, 0, 1);   // next K-step
      __bf16* dst = As + row * 32 + col;
      #pragma unroll
      for (int i = 0; i < 16; i += 4) {
        float4 f = *(const float4*)(src + i);
        dst[i + 0] = (__bf16)f.x; dst[i + 1] = (__bf16)f.y;
        dst[i + 2] = (__bf16)f.z; dst[i + 3] = (__bf16)f.w;
      }
    }
    // --- stage B tile: W[k0..+31][n0..+127] -> Bs[n][k] (transpose in LDS)
    {
      const int kk = tid >> 3;          // 0..31
      const int nn = (tid & 7) * 16;    // 0..112
      const float* src = W + (size_t)(k0 + kk) * Hsz + n0 + nn;
      __builtin_prefetch(src + (size_t)32 * Hsz, 0, 1);  // next K-step
      #pragma unroll
      for (int i = 0; i < 16; i += 4) {
        float4 f = *(const float4*)(src + i);
        Bs[(nn + i + 0) * 32 + kk] = (__bf16)f.x;
        Bs[(nn + i + 1) * 32 + kk] = (__bf16)f.y;
        Bs[(nn + i + 2) * 32 + kk] = (__bf16)f.z;
        Bs[(nn + i + 3) * 32 + kk] = (__bf16)f.w;
      }
    }
    __syncthreads();

    // A fragment for this wave's 16 rows (16-bit A 16x32 layout)
    ABfrag a;
    {
      const __bf16* ap = As + (wave * 16 + l16) * 32;
      a.q[0] = *(const uint4*)(ap + 16 * hl);       // K = 16*hl .. +7
      a.q[1] = *(const uint4*)(ap + 8 + 16 * hl);   // K = 16*hl+8 .. +7
    }
    // preload ALL B fragments, then issue WMMAs back-to-back
    ABfrag bf[8];
    #pragma unroll
    for (int t = 0; t < 8; ++t) {
      const __bf16* bp = Bs + (t * 16 + l16) * 32 + 16 * hl;
      bf[t].q[0] = *(const uint4*)(bp);
      bf[t].q[1] = *(const uint4*)(bp + 8);
    }
    #pragma unroll
    for (int t = 0; t < 8; ++t) acc[t] = bwmma(a.v, bf[t].v, acc[t]);
    __syncthreads();
  }

  // store + bias into [B, NH, S, HD] as bf16
  #pragma unroll
  for (int t = 0; t < 8; ++t) {
    const int n = n0 + t * 16 + l16;
    const float bb = bias[n];
    const int head = n >> 6, d = n & 63;
    #pragma unroll
    for (int r = 0; r < 8; ++r) {
      const int m = m0 + wave * 16 + r + 8 * hl;
      const int bidx = m >> 11, s = m & 2047;
      Out[(((bidx * NHD + head) * Ssz) + s) * HDm + d] = (__bf16)(acc[t][r] + bb);
    }
  }
}

// ---------------------------------------------------------------------------
// Kernel 2: flash attention.  grid = (S/128, NH, B), block = 256 (8 waves).
// Each wave owns 16 query rows; loop over 128-key blocks staged in LDS.
// ---------------------------------------------------------------------------
__global__ __launch_bounds__(256) void attn_kernel(
    const __bf16* __restrict__ Qb, const __bf16* __restrict__ Kb,
    const __bf16* __restrict__ Vb, const float* __restrict__ mask,
    __bf16* __restrict__ Cx)
{
  __shared__ __align__(16) __bf16 Kls[128 * 64];      // [key][d]
  __shared__ __align__(16) __bf16 VT [64 * 128];      // [d][key]
  __shared__ __align__(16) __bf16 Pls[8 * 16 * 128];  // per-wave probs

  const int tid  = threadIdx.x;
  const int wave = tid >> 5;
  const int lane = tid & 31;
  const int hl   = lane >> 4;
  const int l16  = lane & 15;

  const int h = blockIdx.y, b = blockIdx.z;
  const __bf16* Qh = Qb + (size_t)((b * NHD + h) * Ssz) * HDm;
  const __bf16* Kh = Kb + (size_t)((b * NHD + h) * Ssz) * HDm;
  const __bf16* Vh = Vb + (size_t)((b * NHD + h) * Ssz) * HDm;
  const float* mrow = mask + b * Ssz;
  const int q0 = blockIdx.x * 128 + wave * 16;

  // Q fragments (K-dim = HD = 64 -> two 16x32 steps), read directly from global
  ABfrag aq[2];
  {
    const __bf16* qp = Qh + (q0 + l16) * HDm;
    aq[0].q[0] = *(const uint4*)(qp + 16 * hl);
    aq[0].q[1] = *(const uint4*)(qp + 8 + 16 * hl);
    aq[1].q[0] = *(const uint4*)(qp + 32 + 16 * hl);
    aq[1].q[1] = *(const uint4*)(qp + 40 + 16 * hl);
  }

  float rmax[8], rsum[8];
  v8f ctx[4];
  #pragma unroll
  for (int r = 0; r < 8; ++r) { rmax[r] = -1e30f; rsum[r] = 0.f; }
  #pragma unroll
  for (int dt = 0; dt < 4; ++dt) ctx[dt] = vzero();

  for (int j0 = 0; j0 < Ssz; j0 += 128) {
    // stage K block: 128x64 bf16 = 1024 x 16B chunks (async -> LDS if available)
    {
      const __bf16* src = Kh + (size_t)j0 * HDm;
      #pragma unroll
      for (int i = 0; i < 4; ++i) {
        const int c = tid + i * 256;
        cp16_lds(Kls + c * 8, src + c * 8);
      }
    }
    // stage V block transposed into VT[d][key] (register transpose)
    {
      #pragma unroll
      for (int i = 0; i < 4; ++i) {
        const int c = tid + i * 256;       // chunk of 8 elements
        const int key = c >> 3;
        const int d0  = (c & 7) * 8;
        const __bf16* sp = Vh + (size_t)(j0 + key) * HDm + d0;
        #pragma unroll
        for (int e = 0; e < 8; ++e) VT[(d0 + e) * 128 + key] = sp[e];
      }
    }
    cp_wait();
    __syncthreads();

    // scores = Q . K^T  (16x128 per wave = 8 tiles, 2 K-steps each)
    v8f sc[8];
    #pragma unroll
    for (int t = 0; t < 8; ++t) sc[t] = vzero();
    #pragma unroll
    for (int s = 0; s < 2; ++s) {
      #pragma unroll
      for (int g = 0; g < 2; ++g) {   // groups of 4 tiles: batch loads, batch WMMAs
        ABfrag bk[4];
        #pragma unroll
        for (int u = 0; u < 4; ++u) {
          const int t = g * 4 + u;
          const __bf16* kp = Kls + (t * 16 + l16) * 64 + s * 32 + 16 * hl;
          bk[u].q[0] = *(const uint4*)(kp);
          bk[u].q[1] = *(const uint4*)(kp + 8);
        }
        #pragma unroll
        for (int u = 0; u < 4; ++u)
          sc[g * 4 + u] = bwmma(aq[s].v, bk[u].v, sc[g * 4 + u]);
      }
    }
    // scale + additive mask
    #pragma unroll
    for (int t = 0; t < 8; ++t) {
      const float mk = mrow[j0 + t * 16 + l16];
      #pragma unroll
      for (int r = 0; r < 8; ++r) sc[t][r] = sc[t][r] * 0.125f + mk;
    }
    // online softmax: row max / exp / row sum (N striped over 16 lanes)
    #pragma unroll
    for (int r = 0; r < 8; ++r) {
      float mx = sc[0][r];
      #pragma unroll
      for (int t = 1; t < 8; ++t) mx = fmaxf(mx, sc[t][r]);
      #pragma unroll
      for (int o = 8; o >= 1; o >>= 1) mx = fmaxf(mx, __shfl_xor(mx, o, 16));
      const float nm  = fmaxf(rmax[r], mx);
      const float fac = __expf(rmax[r] - nm);
      rmax[r] = nm;
      rsum[r] *= fac;
      #pragma unroll
      for (int dt = 0; dt < 4; ++dt) ctx[dt][r] *= fac;
      float ps = 0.f;
      #pragma unroll
      for (int t = 0; t < 8; ++t) {
        const float p = __expf(sc[t][r] - nm);
        sc[t][r] = p;
        ps += p;
      }
      #pragma unroll
      for (int o = 8; o >= 1; o >>= 1) ps += __shfl_xor(ps, o, 16);
      rsum[r] += ps;
    }
    // probs C-layout -> LDS -> A-layout (per-wave region, DS is in-order)
    __bf16* pw = Pls + wave * 16 * 128;
    #pragma unroll
    for (int t = 0; t < 8; ++t)
      #pragma unroll
      for (int r = 0; r < 8; ++r)
        pw[(r + 8 * hl) * 128 + t * 16 + l16] = (__bf16)sc[t][r];

    // ctx += P . V  (4 K-steps of 32 keys x 4 d-tiles; batch loads per step)
    #pragma unroll
    for (int s = 0; s < 4; ++s) {
      ABfrag ap;
      const __bf16* pp = pw + l16 * 128 + s * 32;
      ap.q[0] = *(const uint4*)(pp + 16 * hl);
      ap.q[1] = *(const uint4*)(pp + 8 + 16 * hl);
      ABfrag bv4[4];
      #pragma unroll
      for (int dt = 0; dt < 4; ++dt) {
        const __bf16* vp = VT + (dt * 16 + l16) * 128 + s * 32 + 16 * hl;
        bv4[dt].q[0] = *(const uint4*)(vp);
        bv4[dt].q[1] = *(const uint4*)(vp + 8);
      }
      #pragma unroll
      for (int dt = 0; dt < 4; ++dt) ctx[dt] = bwmma(ap.v, bv4[dt].v, ctx[dt]);
    }
    __syncthreads();
  }

  // normalize and store ctx as bf16 [B, S, H]
  #pragma unroll
  for (int dt = 0; dt < 4; ++dt)
    #pragma unroll
    for (int r = 0; r < 8; ++r) {
      const int row = q0 + r + 8 * hl;
      const float v = ctx[dt][r] / rsum[r];
      Cx[(size_t)(b * Ssz + row) * Hsz + h * HDm + dt * 16 + l16] = (__bf16)v;
    }
}

// ---------------------------------------------------------------------------
// Kernel 3: output projection.  Tmp = Ctx(bf16) @ Wo + bo  (f32 out)
// ---------------------------------------------------------------------------
__global__ __launch_bounds__(256) void out_gemm(
    const __bf16* __restrict__ Cx, const float* __restrict__ Wo,
    const float* __restrict__ bo, float* __restrict__ Tmp)
{
  __shared__ __align__(16) __bf16 As[128 * 32];
  __shared__ __align__(16) __bf16 Bs[128 * 32];

  const int tid  = threadIdx.x;
  const int wave = tid >> 5;
  const int lane = tid & 31;
  const int hl   = lane >> 4;
  const int l16  = lane & 15;
  const int m0   = blockIdx.x * 128;
  const int n0   = blockIdx.y * 128;

  v8f acc[8];
  #pragma unroll
  for (int t = 0; t < 8; ++t) acc[t] = vzero();

  for (int k0 = 0; k0 < Hsz; k0 += 32) {
    // A tile: already bf16 -> async LDS copy (2 x 16B per thread)
    {
      #pragma unroll
      for (int i = 0; i < 2; ++i) {
        const int c = tid * 2 + i;        // 0..511 chunks of 8
        const int row = c >> 2;
        const int col = (c & 3) * 8;
        cp16_lds(As + row * 32 + col,
                 Cx + (size_t)(m0 + row) * Hsz + k0 + col);
      }
    }
    // B tile: Wo f32 -> bf16 transposed
    {
      const int kk = tid >> 3;
      const int nn = (tid & 7) * 16;
      const float* src = Wo + (size_t)(k0 + kk) * Hsz + n0 + nn;
      __builtin_prefetch(src + (size_t)32 * Hsz, 0, 1);
      #pragma unroll
      for (int i = 0; i < 16; i += 4) {
        float4 f = *(const float4*)(src + i);
        Bs[(nn + i + 0) * 32 + kk] = (__bf16)f.x;
        Bs[(nn + i + 1) * 32 + kk] = (__bf16)f.y;
        Bs[(nn + i + 2) * 32 + kk] = (__bf16)f.z;
        Bs[(nn + i + 3) * 32 + kk] = (__bf16)f.w;
      }
    }
    cp_wait();
    __syncthreads();

    ABfrag a;
    {
      const __bf16* ap = As + (wave * 16 + l16) * 32;
      a.q[0] = *(const uint4*)(ap + 16 * hl);
      a.q[1] = *(const uint4*)(ap + 8 + 16 * hl);
    }
    ABfrag bf[8];
    #pragma unroll
    for (int t = 0; t < 8; ++t) {
      const __bf16* bp = Bs + (t * 16 + l16) * 32 + 16 * hl;
      bf[t].q[0] = *(const uint4*)(bp);
      bf[t].q[1] = *(const uint4*)(bp + 8);
    }
    #pragma unroll
    for (int t = 0; t < 8; ++t) acc[t] = bwmma(a.v, bf[t].v, acc[t]);
    __syncthreads();
  }

  #pragma unroll
  for (int t = 0; t < 8; ++t) {
    const int n = n0 + t * 16 + l16;
    const float bb = bo[n];
    #pragma unroll
    for (int r = 0; r < 8; ++r) {
      const int m = m0 + wave * 16 + r + 8 * hl;
      Tmp[(size_t)m * Hsz + n] = acc[t][r] + bb;
    }
  }
}

// ---------------------------------------------------------------------------
// Kernel 4: residual + LayerNorm.  grid = 8192 rows, block = 256.
// ---------------------------------------------------------------------------
__global__ __launch_bounds__(256) void ln_kernel(
    const float* __restrict__ Tmp, const float* __restrict__ X,
    const float* __restrict__ gamma, const float* __restrict__ beta,
    float* __restrict__ Out)
{
  const int m = blockIdx.x;
  const int tid = threadIdx.x;
  const int wave = tid >> 5, lane = tid & 31;
  __shared__ float w1[8], w2[8];

  float vals[4];
  float s1 = 0.f, s2 = 0.f;
  #pragma unroll
  for (int i = 0; i < 4; ++i) {
    const int c = tid + i * 256;
    const float v = Tmp[(size_t)m * Hsz + c] + X[(size_t)m * Hsz + c];
    vals[i] = v; s1 += v; s2 += v * v;
  }
  #pragma unroll
  for (int o = 16; o >= 1; o >>= 1) {
    s1 += __shfl_xor(s1, o, 32);
    s2 += __shfl_xor(s2, o, 32);
  }
  if (lane == 0) { w1[wave] = s1; w2[wave] = s2; }
  __syncthreads();
  float t1 = 0.f, t2 = 0.f;
  #pragma unroll
  for (int i = 0; i < 8; ++i) { t1 += w1[i]; t2 += w2[i]; }
  const float mean = t1 * (1.f / Hsz);
  const float var  = t2 * (1.f / Hsz) - mean * mean;
  const float inv  = rsqrtf(var + 1e-12f);
  #pragma unroll
  for (int i = 0; i < 4; ++i) {
    const int c = tid + i * 256;
    Out[(size_t)m * Hsz + c] = (vals[i] - mean) * inv * gamma[c] + beta[c];
  }
}

// ---------------------------------------------------------------------------
extern "C" void kernel_launch(void* const* d_in, const int* in_sizes, int n_in,
                              void* d_out, int out_size, void* d_ws, size_t ws_size,
                              hipStream_t stream) {
  (void)in_sizes; (void)n_in; (void)out_size; (void)ws_size;
  const float* hid  = (const float*)d_in[0];
  const float* mask = (const float*)d_in[1];
  const float* Wq = (const float*)d_in[2];
  const float* bq = (const float*)d_in[3];
  const float* Wk = (const float*)d_in[4];
  const float* bk = (const float*)d_in[5];
  const float* Wv = (const float*)d_in[6];
  const float* bv = (const float*)d_in[7];
  const float* Wo = (const float*)d_in[8];
  const float* bo = (const float*)d_in[9];
  const float* gamma = (const float*)d_in[10];
  const float* beta  = (const float*)d_in[11];
  float* out = (float*)d_out;

  const size_t elems = (size_t)Mrows * Hsz;          // 8M elements
  __bf16* Qb = (__bf16*)d_ws;                        // 16 MiB
  __bf16* Kb = Qb + elems;                           // 16 MiB
  __bf16* Vb = Kb + elems;                           // 16 MiB
  __bf16* Cx = Vb + elems;                           // 16 MiB
  float*  Tmp = (float*)(Cx + elems);                // 32 MiB

  qkv_gemm<<<dim3(Mrows / 128, Hsz / 128, 3), 256, 0, stream>>>(
      hid, Wq, bq, Wk, bk, Wv, bv, Qb, Kb, Vb);
  attn_kernel<<<dim3(Ssz / 128, NHD, Bsz), 256, 0, stream>>>(
      Qb, Kb, Vb, mask, Cx);
  out_gemm<<<dim3(Mrows / 128, Hsz / 128, 1), 256, 0, stream>>>(
      Cx, Wo, bo, Tmp);
  ln_kernel<<<dim3(Mrows), 256, 0, stream>>>(Tmp, hid, gamma, beta, out);
}
